// GroupedQueryAttention_37864431681755
// MI455X (gfx1250) — matrix operand order repaired and checked by reading, not compiled, not run on hardware
//
#include <hip/hip_runtime.h>
#include <math.h>

// ---------------------------------------------------------------------------
// GQA (degenerate per-position softmax over G=4 groups).
// GEMMs (~344 GFLOP) dominate; memory is ~25us at 23.3 TB/s => compute-bound.
// fp32 matrix rate on CDNA is ~1/8 of fp16, so run the GEMMs as fp16x3 split
// (Ah*Bh + Ah*Bl + Al*Bh, f32 accumulate) on v_wmma_f32_16x16x32_f16:
// fp32-grade accuracy (~1e-5 rel), ~2.7x the fp32-WMMA throughput.
// ---------------------------------------------------------------------------

typedef __attribute__((ext_vector_type(8)))  float    v8f;
typedef __attribute__((ext_vector_type(16))) _Float16 v16h;
typedef __attribute__((ext_vector_type(8)))  _Float16 h8;
typedef __attribute__((ext_vector_type(4)))  _Float16 h4;

#define BM 128
#define BN 128
#define BK 32
#define KP 40   // LDS row stride in halves: 80B rows -> conflict-free b128 reads

// C[M,N] = A[M,K] @ B[K,N] + bias[N] ; fp32 in/out, fp16x3 internally.
// Requires M%128==0, N%128==0, K%32==0 (true for all four GEMMs here).
__global__ __launch_bounds__(256)
void gemm_f16x3_wmma(const float* __restrict__ A, const float* __restrict__ B,
                     const float* __restrict__ bias, float* __restrict__ C,
                     int M, int N, int K) {
  __shared__ _Float16 Ah[BM * KP];
  __shared__ _Float16 Al[BM * KP];
  __shared__ _Float16 Bh[BN * KP];   // stored K-major: Bh[n][k]
  __shared__ _Float16 Bl[BN * KP];

  const int tid  = threadIdx.x;
  const int lane = tid & 31;
  const int wave = tid >> 5;      // 0..7
  const int lo   = lane & 15;
  const int hi   = lane >> 4;     // 0/1 : selects K-half per the 16-bit layouts
  const int m0   = (wave >> 1) * 32;  // wave M strip
  const int n0   = (wave & 1) * 64;   // wave N strip
  const int bm   = blockIdx.y * BM;
  const int bn   = blockIdx.x * BN;

  // Staging thread mapping
  const int ncol  = tid & 127;        // B: column within tile
  const int khalf = (tid >> 7) * 16;  // B: k-range 0..15 or 16..31

  v8f acc[2][4];
#pragma unroll
  for (int mi = 0; mi < 2; ++mi)
#pragma unroll
    for (int ni = 0; ni < 4; ++ni) acc[mi][ni] = (v8f)0.0f;

  // ---- register staging (prefetch tile kb=0) ----
  float4 areg[4];
  float  breg[16];
#pragma unroll
  for (int i = 0; i < 4; ++i) {
    const int idx = tid + 256 * i;
    const int r = idx >> 3, c = (idx & 7) << 2;   // A tile 128 x 32
    areg[i] = *(const float4*)(A + (size_t)(bm + r) * K + c);
  }
#pragma unroll
  for (int kk = 0; kk < 16; ++kk)                  // B columns: coalesced rows
    breg[kk] = B[(size_t)(khalf + kk) * N + bn + ncol];

  for (int kb = 0; kb < K; kb += BK) {
    __syncthreads();  // previous tile fully consumed

    // ---- split f32 -> (hi,lo) f16 and store to LDS ----
#pragma unroll
    for (int i = 0; i < 4; ++i) {
      const int idx = tid + 256 * i;
      const int r = idx >> 3, c = (idx & 7) << 2;
      float v[4] = {areg[i].x, areg[i].y, areg[i].z, areg[i].w};
      h4 hh, ll;
#pragma unroll
      for (int j = 0; j < 4; ++j) {
        _Float16 h = (_Float16)v[j];
        hh[j] = h;
        ll[j] = (_Float16)(v[j] - (float)h);
      }
      *(h4*)(&Ah[r * KP + c]) = hh;
      *(h4*)(&Al[r * KP + c]) = ll;
    }
    {
      h8 bh0, bh1, bl0, bl1;
#pragma unroll
      for (int kk = 0; kk < 8; ++kk) {
        _Float16 h0 = (_Float16)breg[kk];
        _Float16 h1 = (_Float16)breg[kk + 8];
        bh0[kk] = h0; bl0[kk] = (_Float16)(breg[kk] - (float)h0);
        bh1[kk] = h1; bl1[kk] = (_Float16)(breg[kk + 8] - (float)h1);
      }
      *(h8*)(&Bh[ncol * KP + khalf])     = bh0;
      *(h8*)(&Bh[ncol * KP + khalf + 8]) = bh1;
      *(h8*)(&Bl[ncol * KP + khalf])     = bl0;
      *(h8*)(&Bl[ncol * KP + khalf + 8]) = bl1;
    }
    __syncthreads();

    // ---- prefetch next tile while computing (uniform branch) ----
    const int kn = kb + BK;
    if (kn < K) {
#pragma unroll
      for (int i = 0; i < 4; ++i) {
        const int idx = tid + 256 * i;
        const int r = idx >> 3, c = (idx & 7) << 2;
        areg[i] = *(const float4*)(A + (size_t)(bm + r) * K + kn + c);
      }
#pragma unroll
      for (int kk = 0; kk < 16; ++kk)
        breg[kk] = B[(size_t)(kn + khalf + kk) * N + bn + ncol];
    }

    // ---- fragments + 24 WMMA ----
    // A frag (16x32 f16): lane m=lo, half j -> K = 16*(j>=8) + 8*hi + (j&7)
    // B frag (32x16 f16): lane n=lo, half j -> K = 16*(j>=8) + 8*hi + (j&7)  (K-major in LDS)
    v16h afh[2], afl[2];
#pragma unroll
    for (int mi = 0; mi < 2; ++mi) {
      const int row = m0 + mi * 16 + lo;
      h8 c0 = *(const h8*)(&Ah[row * KP + 8 * hi]);
      h8 c1 = *(const h8*)(&Ah[row * KP + 16 + 8 * hi]);
      afh[mi] = __builtin_shufflevector(c0, c1, 0, 1, 2, 3, 4, 5, 6, 7, 8, 9,
                                        10, 11, 12, 13, 14, 15);
      h8 d0 = *(const h8*)(&Al[row * KP + 8 * hi]);
      h8 d1 = *(const h8*)(&Al[row * KP + 16 + 8 * hi]);
      afl[mi] = __builtin_shufflevector(d0, d1, 0, 1, 2, 3, 4, 5, 6, 7, 8, 9,
                                        10, 11, 12, 13, 14, 15);
    }
#pragma unroll
    for (int ni = 0; ni < 4; ++ni) {
      const int col = n0 + ni * 16 + lo;
      h8 b0 = *(const h8*)(&Bh[col * KP + 8 * hi]);
      h8 b1 = *(const h8*)(&Bh[col * KP + 16 + 8 * hi]);
      v16h bfh = __builtin_shufflevector(b0, b1, 0, 1, 2, 3, 4, 5, 6, 7, 8, 9,
                                         10, 11, 12, 13, 14, 15);
      h8 e0 = *(const h8*)(&Bl[col * KP + 8 * hi]);
      h8 e1 = *(const h8*)(&Bl[col * KP + 16 + 8 * hi]);
      v16h bfl = __builtin_shufflevector(e0, e1, 0, 1, 2, 3, 4, 5, 6, 7, 8, 9,
                                         10, 11, 12, 13, 14, 15);
#pragma unroll
      for (int mi = 0; mi < 2; ++mi) {
        acc[mi][ni] = __builtin_amdgcn_wmma_f32_16x16x32_f16(
            false, afh[mi], false, bfh, (short)0, acc[mi][ni], false, false);
        acc[mi][ni] = __builtin_amdgcn_wmma_f32_16x16x32_f16(
            false, afh[mi], false, bfl, (short)0, acc[mi][ni], false, false);
        acc[mi][ni] = __builtin_amdgcn_wmma_f32_16x16x32_f16(
            false, afl[mi], false, bfh, (short)0, acc[mi][ni], false, false);
      }
    }
  }

  // Epilogue: C/D layout: comp r -> row (r + 8*hi), col = lane&15
#pragma unroll
  for (int mi = 0; mi < 2; ++mi)
#pragma unroll
    for (int ni = 0; ni < 4; ++ni) {
      const int n = bn + n0 + ni * 16 + lo;
      const float bb = bias[n];
#pragma unroll
      for (int r = 0; r < 8; ++r) {
        const int m = bm + m0 + mi * 16 + r + 8 * hi;
        C[(size_t)m * N + n] = acc[mi][ni][r] + bb;
      }
    }
}

// One block per token; 8 waves -> 2 heads each; lane owns dims [4*lane,4*lane+3].
// Writes output in place over Q (safe: wave reads its q slice before writing).
__global__ __launch_bounds__(256)
void gqa_mix_kernel(float* __restrict__ Q, const float* __restrict__ K,
                    const float* __restrict__ V) {
  const int tok  = blockIdx.x;
  const int lane = threadIdx.x & 31;
  const int wave = threadIdx.x >> 5;
  const float scale = 0.08838834764831845f;  // 1/sqrt(128)

  const float4* Kt = (const float4*)(K + (size_t)tok * 512);
  const float4* Vt = (const float4*)(V + (size_t)tok * 512);
  float4 kf[4], vf[4];
#pragma unroll
  for (int g = 0; g < 4; ++g) {
    kf[g] = Kt[g * 32 + lane];
    vf[g] = Vt[g * 32 + lane];
  }

  float4* Qt = (float4*)(Q + (size_t)tok * 2048);
#pragma unroll
  for (int hh = 0; hh < 2; ++hh) {
    const int h = wave * 2 + hh;
    const float4 qf = Qt[h * 32 + lane];
    float s[4];
#pragma unroll
    for (int g = 0; g < 4; ++g) {
      float p = qf.x * kf[g].x + qf.y * kf[g].y + qf.z * kf[g].z + qf.w * kf[g].w;
#pragma unroll
      for (int off = 16; off > 0; off >>= 1) p += __shfl_xor(p, off, 32);
      s[g] = p * scale;
    }
    float mx = fmaxf(fmaxf(s[0], s[1]), fmaxf(s[2], s[3]));
    float pr[4], sum = 0.0f;
#pragma unroll
    for (int g = 0; g < 4; ++g) { pr[g] = __expf(s[g] - mx); sum += pr[g]; }
    const float inv = 1.0f / sum;
    float4 o = make_float4(0.f, 0.f, 0.f, 0.f);
#pragma unroll
    for (int g = 0; g < 4; ++g) {
      const float w = pr[g] * inv;
      o.x += w * vf[g].x; o.y += w * vf[g].y; o.z += w * vf[g].z; o.w += w * vf[g].w;
    }
    Qt[h * 32 + lane] = o;
  }
}

extern "C" void kernel_launch(void* const* d_in, const int* in_sizes, int n_in,
                              void* d_out, int out_size, void* d_ws, size_t ws_size,
                              hipStream_t stream) {
  (void)in_sizes; (void)n_in; (void)out_size; (void)ws_size;
  const float* x  = (const float*)d_in[0];
  const float* Wq = (const float*)d_in[1];
  const float* bq = (const float*)d_in[2];
  const float* Wk = (const float*)d_in[3];
  const float* bk = (const float*)d_in[4];
  const float* Wv = (const float*)d_in[5];
  const float* bv = (const float*)d_in[6];
  const float* Wo = (const float*)d_in[7];
  const float* bo = (const float*)d_in[8];
  float* out = (float*)d_out;

  const int Mtok = 4 * 4096;  // 16384 tokens
  const int HID = 2048, KV = 512;

  float* q = (float*)d_ws;                      // 16384*2048 f32
  float* k = q + (size_t)Mtok * HID;            // 16384*512
  float* v = k + (size_t)Mtok * KV;             // 16384*512  (~201 MB total)

  dim3 blk(256);
  gemm_f16x3_wmma<<<dim3(HID / BN, Mtok / BM), blk, 0, stream>>>(x, Wq, bq, q, Mtok, HID, HID);
  gemm_f16x3_wmma<<<dim3(KV  / BN, Mtok / BM), blk, 0, stream>>>(x, Wk, bk, k, Mtok, KV,  HID);
  gemm_f16x3_wmma<<<dim3(KV  / BN, Mtok / BM), blk, 0, stream>>>(x, Wv, bv, v, Mtok, KV,  HID);
  gqa_mix_kernel<<<dim3(Mtok), blk, 0, stream>>>(q, k, v);
  gemm_f16x3_wmma<<<dim3(HID / BN, Mtok / BM), blk, 0, stream>>>(q, Wo, bo, out, Mtok, HID, HID);
}
